// NormLinearAttention_26182120636679
// MI455X (gfx1250) — matrix operand order repaired and checked by reading, not compiled, not run on hardware
//
#include <hip/hip_runtime.h>

// ---------------------------------------------------------------------------
// NormLinearAttention for MI455X (gfx1250, wave32, WMMA bf16, async-LDS copy)
// ---------------------------------------------------------------------------

typedef __attribute__((ext_vector_type(16))) __bf16 v16bf;
typedef __attribute__((ext_vector_type(8)))  __bf16 v8bf;
typedef __attribute__((ext_vector_type(8)))  float  v8f;

#define BM 128
#define BN 128
#define BK 32

// ---- gfx1250 async global->LDS copy helpers (ASYNCcnt-tracked) ------------
__device__ __forceinline__ unsigned lds_addr_of(const void* p)
{
    // generic -> LDS(addrspace 3) -> 32-bit LDS byte offset
    return (unsigned)(uintptr_t)(__attribute__((address_space(3))) const void*)p;
}

__device__ __forceinline__ void async_copy_b128(unsigned lds_byte, const void* gptr)
{
    asm volatile("global_load_async_to_lds_b128 %0, %1, off"
                 :: "v"(lds_byte), "v"((unsigned long long)(uintptr_t)gptr)
                 : "memory");
}

#define WAIT_ASYNC_LE4() asm volatile("s_wait_asynccnt 0x4" ::: "memory")
#define WAIT_ASYNC_0()   asm volatile("s_wait_asynccnt 0x0" ::: "memory")

// ---------------------------------------------------------------------------
// Shared WMMA GEMM:  C = act(A @ Bt^T + bias)
//   A : bf16, [M x K], row stride lda, with per-z offsets  z=(b,h)
//   Bt: bf16, [N x K]  (pre-transposed weights / kv), per-z stride btStrideZ
//   C : OutT, row stride ldc, per-z offsets
//   act: 0 = none, 1 = elu, 2 = silu
// Block = 256 threads = 8 wave32. Wave w computes C rows [16w,16w+16) x BN.
// Double-buffered LDS tiles staged with global_load_async_to_lds_b128 so the
// next tile's HBM traffic overlaps the current tile's WMMAs.
// ---------------------------------------------------------------------------
template <typename OutT>
__global__ void __launch_bounds__(256)
wmma_gemm(const __bf16* __restrict__ A, long lda, long aStrideB, long aStrideH,
          const __bf16* __restrict__ Bt, long btStrideZ,
          const float* __restrict__ bias,
          OutT* __restrict__ C, long ldc, long cStrideB, long cStrideH,
          int M, int N, int K, int heads, int act)
{
    __shared__ __attribute__((aligned(32))) __bf16 lA[2][BM][BK];
    __shared__ __attribute__((aligned(32))) __bf16 lB[2][BN][BK];

    const int z  = blockIdx.z;
    const int bb = z / heads;
    const int hh = z % heads;
    A  += (long)bb * aStrideB + (long)hh * aStrideH;
    Bt += (long)z * btStrideZ;
    C  += (long)bb * cStrideB + (long)hh * cStrideH;

    const int tid  = threadIdx.x;
    const int wave = tid >> 5;
    const int lane = tid & 31;
    const int half = lane >> 4;   // which 16-lane half
    const int l16  = lane & 15;

    const int m0 = blockIdx.y * BM;
    const int n0 = blockIdx.x * BN;

    v8f acc[8] = {};

    // issue 4 async b128 copies per thread per stage (A: 2, Bt: 2)
    auto stage = [&](int buf, int k0) {
        for (int p = 0; p < 2; ++p) {
            const int id = tid + p * 256;
            const int r  = id >> 2;
            const int c  = (id & 3) << 3;
            async_copy_b128(lds_addr_of(&lA[buf][r][c]),
                            &A[(long)(m0 + r) * lda + k0 + c]);
        }
        for (int p = 0; p < 2; ++p) {
            const int id = tid + p * 256;
            const int r  = id >> 2;
            const int c  = (id & 3) << 3;
            async_copy_b128(lds_addr_of(&lB[buf][r][c]),
                            &Bt[(long)(n0 + r) * (long)K + k0 + c]);
        }
    };

    const int nk = K / BK;
    stage(0, 0);

    for (int t = 0; t < nk; ++t) {
        const int buf = t & 1;
        if (t + 1 < nk) {
            stage(buf ^ 1, (t + 1) * BK);   // overlap next tile with compute
            WAIT_ASYNC_LE4();               // previous stage (4/thread) landed
        } else {
            WAIT_ASYNC_0();
        }
        __syncthreads();

        // L2 prefetch two tiles ahead (global_prefetch_b8)
        if (t + 2 < nk)
            __builtin_prefetch(&A[(long)(m0 + (tid >> 1)) * lda + (t + 2) * BK], 0, 1);

        // ---- A fragment: row m = 16*wave + l16, contiguous k = 16*half + s ----
        v16bf afrag;
        {
            const int m = 16 * wave + l16;
            afrag = *reinterpret_cast<const v16bf*>(&lA[buf][m][half * 16]);
        }

        // ---- 8 WMMAs across the N tiles ----
        for (int j = 0; j < 8; ++j) {
            const int n = 16 * j + l16;
            v16bf bfrag = *reinterpret_cast<const v16bf*>(&lB[buf][n][half * 16]);
            acc[j] = __builtin_amdgcn_wmma_f32_16x16x32_bf16(
                false, afrag, false, bfrag, (short)0, acc[j], false, false);
        }
        __syncthreads();   // all waves done reading buf before it is restaged
    }

    // ---- epilogue: bias + activation, D layout m = r + 8*half, n = l16 ----
    for (int j = 0; j < 8; ++j) {
        const int n  = n0 + 16 * j + l16;
        const float bn = bias ? bias[n] : 0.0f;
        for (int r = 0; r < 8; ++r) {
            const int m = m0 + 16 * wave + r + 8 * half;
            float val = acc[j][r] + bn;
            if (act == 1)      val = val > 0.0f ? val : (__expf(val) - 1.0f);   // elu
            else if (act == 2) val = val / (1.0f + __expf(-val));               // silu
            C[(long)m * ldc + n] = (OutT)val;
        }
    }
}

// ---------------------------------------------------------------------------
// kv partial: kv[bh][d][e] += sum_{n in chunk} qk[b,n,h,d] * v[b,n,h,e]
// grid = (64 bh, 4 n-chunks); transposed LDS staging so both WMMA fragments
// read contiguous-K; fp32 global atomic reduction across the 4 chunks.
// ---------------------------------------------------------------------------
__global__ void __launch_bounds__(256)
kv_partial(const __bf16* __restrict__ qk, const __bf16* __restrict__ v,
           float* __restrict__ kvf)
{
    __shared__ __attribute__((aligned(32))) __bf16 lQ[128][BK];  // [d][n]
    __shared__ __attribute__((aligned(32))) __bf16 lV[128][BK];  // [e][n]

    const int bh = blockIdx.x;            // 0..63
    const int b  = bh >> 4, h = bh & 15;
    const long rowBase = (long)b * 4096;

    const int tid  = threadIdx.x;
    const int wave = tid >> 5;
    const int lane = tid & 31;
    const int half = lane >> 4;
    const int l16  = lane & 15;

    v8f acc[8] = {};

    const int nBeg = blockIdx.y * 1024;
    for (int n0 = nBeg; n0 < nBeg + 1024; n0 += BK) {
        // load [32 n x 128 c] rows of qk / v for head h, store transposed
        {
            const int nl = tid >> 3;          // 0..31
            const int c0 = (tid & 7) << 4;    // 0,16,...,112
            const __bf16* qrow = &qk[(rowBase + n0 + nl) * 2048 + h * 128 + c0];
            const __bf16* vrow = &v [(rowBase + n0 + nl) * 2048 + h * 128 + c0];
            v8bf q0 = *reinterpret_cast<const v8bf*>(qrow);
            v8bf q1 = *reinterpret_cast<const v8bf*>(qrow + 8);
            v8bf w0 = *reinterpret_cast<const v8bf*>(vrow);
            v8bf w1 = *reinterpret_cast<const v8bf*>(vrow + 8);
            for (int e = 0; e < 8; ++e) {
                lQ[c0 + e][nl]     = q0[e];
                lQ[c0 + 8 + e][nl] = q1[e];
                lV[c0 + e][nl]     = w0[e];
                lV[c0 + 8 + e][nl] = w1[e];
            }
        }
        __syncthreads();

        v16bf afrag;   // A[m=d][k=n] = qk[n][d]
        {
            const int d = 16 * wave + l16;
            afrag = *reinterpret_cast<const v16bf*>(&lQ[d][half * 16]);
        }
        for (int j = 0; j < 8; ++j) {
            const int e = 16 * j + l16;    // B[k=n][nn=e] = v[n][e]
            v16bf bfrag = *reinterpret_cast<const v16bf*>(&lV[e][half * 16]);
            acc[j] = __builtin_amdgcn_wmma_f32_16x16x32_bf16(
                false, afrag, false, bfrag, (short)0, acc[j], false, false);
        }
        __syncthreads();
    }

    float* out = kvf + (long)bh * 128 * 128;
    for (int j = 0; j < 8; ++j) {
        const int e = 16 * j + l16;
        for (int r = 0; r < 8; ++r) {
            const int d = 16 * wave + r + 8 * half;
            atomicAdd(&out[d * 128 + e], acc[j][r]);
        }
    }
}

// ---------------------------------------------------------------------------
// small helper kernels
// ---------------------------------------------------------------------------
__global__ void __launch_bounds__(256)
f32_to_bf16_k(const float* __restrict__ in, __bf16* __restrict__ out, long n)
{
    long i = (long)blockIdx.x * blockDim.x + threadIdx.x;
    const long stride = (long)gridDim.x * blockDim.x;
    for (; i < n; i += stride) out[i] = (__bf16)in[i];
}

__global__ void __launch_bounds__(256)
transpose_bf16_k(const float* __restrict__ W, __bf16* __restrict__ Wt, int K, int N)
{
    long i = (long)blockIdx.x * blockDim.x + threadIdx.x;
    const long stride = (long)gridDim.x * blockDim.x;
    const long total = (long)K * N;
    for (; i < total; i += stride) {
        const int k = (int)(i / N);
        const int n = (int)(i % N);
        Wt[(long)n * K + k] = (__bf16)W[i];
    }
}

__global__ void __launch_bounds__(256)
zero_f32_k(float* __restrict__ p, long n)
{
    long i = (long)blockIdx.x * blockDim.x + threadIdx.x;
    const long stride = (long)gridDim.x * blockDim.x;
    for (; i < n; i += stride) p[i] = 0.0f;
}

__global__ void __launch_bounds__(256)
kv_cvt_k(const float* __restrict__ kvf, __bf16* __restrict__ kvT)
{
    long i = (long)blockIdx.x * blockDim.x + threadIdx.x;
    const long stride = (long)gridDim.x * blockDim.x;
    const long total = 64L * 128 * 128;
    for (; i < total; i += stride) {
        const long bh = i >> 14;
        const int  d  = (int)((i >> 7) & 127);
        const int  e  = (int)(i & 127);
        kvT[(bh << 14) + (long)e * 128 + d] = (__bf16)kvf[i];  // kvT[bh][e][d]
    }
}

// LayerNorm over 2048 fused with u * (.) ; one block (8 wave32) per token row
__global__ void __launch_bounds__(256)
ln_mul_k(const float* __restrict__ outp, const __bf16* __restrict__ u,
         const float* __restrict__ g, const float* __restrict__ bl,
         __bf16* __restrict__ y)
{
    const long row = blockIdx.x;
    const float* xr = outp + row * 2048;
    __shared__ float rs[8], rs2[8];

    float vals[8];
    float s = 0.0f, s2 = 0.0f;
    for (int i = 0; i < 8; ++i) {
        const float t = xr[threadIdx.x + i * 256];
        vals[i] = t; s += t; s2 += t * t;
    }
    for (int off = 16; off > 0; off >>= 1) {
        s  += __shfl_down(s,  off, 32);
        s2 += __shfl_down(s2, off, 32);
    }
    if ((threadIdx.x & 31) == 0) {
        rs[threadIdx.x >> 5]  = s;
        rs2[threadIdx.x >> 5] = s2;
    }
    __syncthreads();
    if (threadIdx.x == 0) {
        float a = 0.0f, b2 = 0.0f;
        for (int w = 0; w < 8; ++w) { a += rs[w]; b2 += rs2[w]; }
        rs[0] = a; rs2[0] = b2;
    }
    __syncthreads();
    const float mean = rs[0] * (1.0f / 2048.0f);
    const float var  = rs2[0] * (1.0f / 2048.0f) - mean * mean;
    const float inv  = rsqrtf(var + 1e-5f);
    for (int i = 0; i < 8; ++i) {
        const int c = threadIdx.x + i * 256;
        float yv = (vals[i] - mean) * inv * g[c] + bl[c];
        yv *= (float)u[row * 2048 + c];
        y[row * 2048 + c] = (__bf16)yv;
    }
}

// ---------------------------------------------------------------------------
// launch
// ---------------------------------------------------------------------------
extern "C" void kernel_launch(void* const* d_in, const int* in_sizes, int n_in,
                              void* d_out, int out_size, void* d_ws, size_t ws_size,
                              hipStream_t stream)
{
    (void)in_sizes; (void)n_in; (void)out_size; (void)ws_size;
    const float* x   = (const float*)d_in[0];
    const float* Wqk = (const float*)d_in[1];
    const float* bqk = (const float*)d_in[2];
    const float* Wv  = (const float*)d_in[3];
    const float* bv  = (const float*)d_in[4];
    const float* Wu  = (const float*)d_in[5];
    const float* bu  = (const float*)d_in[6];
    const float* Wo  = (const float*)d_in[7];
    const float* bo  = (const float*)d_in[8];
    const float* lng = (const float*)d_in[9];
    const float* lnb = (const float*)d_in[10];

    const long T = 16384;   // B*N tokens
    const long E = 2048;
    const long H = 2048;

    char* ws = (char*)d_ws;
    size_t off = 0;
    auto alloc = [&](size_t bytes) -> char* {
        char* p = ws + off;
        off += (bytes + 255) & ~(size_t)255;
        return p;
    };
    __bf16* xb   = (__bf16*)alloc(T * E * 2);
    __bf16* wqkT = (__bf16*)alloc(E * H * 2);
    __bf16* wvT  = (__bf16*)alloc(E * H * 2);
    __bf16* wuT  = (__bf16*)alloc(E * H * 2);
    __bf16* woT  = (__bf16*)alloc(H * E * 2);
    __bf16* qk   = (__bf16*)alloc(T * H * 2);
    __bf16* vv   = (__bf16*)alloc(T * H * 2);
    __bf16* uu   = (__bf16*)alloc(T * H * 2);
    float*  kvf  = (float*)alloc(64L * 128 * 128 * 4);
    __bf16* kvT  = (__bf16*)alloc(64L * 128 * 128 * 2);
    float*  outp = (float*)alloc(T * H * 4);
    __bf16* yln  = (__bf16*)alloc(T * H * 2);

    const dim3 blk(256);

    // 0) precision conversion + weight transposes
    f32_to_bf16_k<<<4096, blk, 0, stream>>>(x, xb, T * E);
    transpose_bf16_k<<<4096, blk, 0, stream>>>(Wqk, wqkT, 2048, 2048);
    transpose_bf16_k<<<4096, blk, 0, stream>>>(Wv,  wvT,  2048, 2048);
    transpose_bf16_k<<<4096, blk, 0, stream>>>(Wu,  wuT,  2048, 2048);
    transpose_bf16_k<<<4096, blk, 0, stream>>>(Wo,  woT,  2048, 2048);

    // 1) projections with fused bias + activation (q==k, so computed once)
    const dim3 gProj(16, 128, 1);
    wmma_gemm<__bf16><<<gProj, blk, 0, stream>>>(xb, 2048, 0, 0, wqkT, 0, bqk,
                                                 qk, 2048, 0, 0, 16384, 2048, 2048, 1, /*elu*/1);
    wmma_gemm<__bf16><<<gProj, blk, 0, stream>>>(xb, 2048, 0, 0, wvT, 0, bv,
                                                 vv, 2048, 0, 0, 16384, 2048, 2048, 1, /*silu*/2);
    wmma_gemm<__bf16><<<gProj, blk, 0, stream>>>(xb, 2048, 0, 0, wuT, 0, bu,
                                                 uu, 2048, 0, 0, 16384, 2048, 2048, 1, /*silu*/2);

    // 2) kv = K^T V per (b,h), split over n with fp32 atomic reduction
    zero_f32_k<<<1024, blk, 0, stream>>>(kvf, 64L * 128 * 128);
    kv_partial<<<dim3(64, 4), blk, 0, stream>>>(qk, vv, kvf);
    kv_cvt_k<<<1024, blk, 0, stream>>>(kvf, kvT);

    // 3) out_pre = Q @ kv per (b,h)
    wmma_gemm<float><<<dim3(1, 32, 64), blk, 0, stream>>>(
        qk, 2048, 4096L * 2048, 128,      // A: per-b / per-h offsets
        kvT, 128L * 128, nullptr,
        outp, 2048, 4096L * 2048, 128,
        4096, 128, 128, /*heads*/16, /*act*/0);

    // 4) LayerNorm(2048) fused with u ⊙ (.)
    ln_mul_k<<<16384, blk, 0, stream>>>(outp, uu, lng, lnb, yln);

    // 5) final projection with bias -> fp32 output
    wmma_gemm<float><<<gProj, blk, 0, stream>>>(yln, 2048, 0, 0, woT, 0, bo,
                                                (float*)d_out, 2048, 0, 0,
                                                16384, 2048, 2048, 1, /*act*/0);
}
